// SetConv_81140522156684
// MI455X (gfx1250) — matrix operand order repaired and checked by reading, not compiled, not run on hardware
//
#include <hip/hip_runtime.h>

// ---------------- problem constants ----------------
#define BATCH 4
#define NPTS  8192
#define CIN   64
#define SOUT  2048
#define KNBR  32
#define RAD2  0.25f
#define BNEPS 1e-3f

// ---------------- CDNA5 WMMA types ----------------
typedef _Float16 v16h  __attribute__((ext_vector_type(16)));
typedef _Float16 half8 __attribute__((ext_vector_type(8)));
typedef float    v8f   __attribute__((ext_vector_type(8)));

union AV { v16h v; half8 h[2]; };

__device__ __forceinline__ v8f wmma_f16(const v16h a, const v16h b, v8f c) {
  // D = A(16x32 f16) * B(32x16 f16) + C(16x16 f32)
  return __builtin_amdgcn_wmma_f32_16x16x32_f16(
      /*neg_a=*/false, a, /*neg_b=*/false, b,
      /*c_mod=*/(short)0, c, /*reuse_a=*/false, /*reuse_b=*/false);
}

// B-fragment: lane L holds column n=L%16, element e <-> K = (L/16)*8 + (e&7) + (e>>3)*16
// blob layout: half index = ((kc*NT + nt)*32 + lane)*16 + e  -> one 32B load/lane
__device__ __forceinline__ v16h loadB(const _Float16* wb, int kc, int NT, int nt, int lane) {
  return *(const v16h*)(wb + (size_t)(((kc * NT + nt) * 32 + lane)) * 16);
}

// ============================================================
// 1) Farthest point sampling: one block per batch, all state in VGPRs
// ============================================================
__global__ __launch_bounds__(1024) void fps_kernel(const float* __restrict__ points,
                                                   int* __restrict__ fpsIdx,
                                                   float* __restrict__ outPts) {
  const int b = blockIdx.x;
  const int tid = threadIdx.x;
  const float* pb = points + (size_t)b * 3 * NPTS;

  float px[8], py[8], pz[8], dd[8];
#pragma unroll
  for (int i = 0; i < 8; ++i) {
    int n = tid + i * 1024;
    px[i] = pb[n];
    py[i] = pb[NPTS + n];
    pz[i] = pb[2 * NPTS + n];
    dd[i] = 1e10f;
  }

  __shared__ float rd[32];
  __shared__ int   rn[32];
  __shared__ int   lastSh;

  int last = 0;
  for (int s = 0; s < SOUT; ++s) {
    float lx = pb[last], ly = pb[NPTS + last], lz = pb[2 * NPTS + last];
    if (tid == 0) {
      fpsIdx[b * SOUT + s] = last;
      outPts[(b * 3 + 0) * SOUT + s] = lx;
      outPts[(b * 3 + 1) * SOUT + s] = ly;
      outPts[(b * 3 + 2) * SOUT + s] = lz;
    }
    float bestd = -1.0f;
    int bestn = 0;
#pragma unroll
    for (int i = 0; i < 8; ++i) {
      float dx = px[i] - lx, dy = py[i] - ly, dz = pz[i] - lz;
      float nd = dx * dx + dy * dy + dz * dz;
      dd[i] = fminf(dd[i], nd);
      if (dd[i] > bestd) { bestd = dd[i]; bestn = tid + i * 1024; }
    }
    // wave32 reduction (argmax, ties -> smaller index like jnp.argmax)
#pragma unroll
    for (int off = 16; off > 0; off >>= 1) {
      float od = __shfl_down(bestd, off);
      int   on = __shfl_down(bestn, off);
      if (od > bestd || (od == bestd && on < bestn)) { bestd = od; bestn = on; }
    }
    if ((tid & 31) == 0) { rd[tid >> 5] = bestd; rn[tid >> 5] = bestn; }
    __syncthreads();
    if (tid < 32) {
      bestd = rd[tid]; bestn = rn[tid];
#pragma unroll
      for (int off = 16; off > 0; off >>= 1) {
        float od = __shfl_down(bestd, off);
        int   on = __shfl_down(bestn, off);
        if (od > bestd || (od == bestd && on < bestn)) { bestd = od; bestn = on; }
      }
      if (tid == 0) lastSh = bestn;
    }
    __syncthreads();
    last = lastSh;
  }
}

// ============================================================
// 2) Ball query: first K in-ball indices (ascending), pad with first hit
// ============================================================
__global__ __launch_bounds__(256) void ball_kernel(const float* __restrict__ points,
                                                   const int* __restrict__ fpsIdx,
                                                   int* __restrict__ nn) {
  const int q = blockIdx.x * blockDim.x + threadIdx.x; // 0 .. B*S-1
  const int b = q >> 11;
  const float* pb = points + (size_t)b * 3 * NPTS;
  const int ci = fpsIdx[q];
  const float qx = pb[ci], qy = pb[NPTS + ci], qz = pb[2 * NPTS + ci];
  int* out = nn + (size_t)q * KNBR;
  int cnt = 0, first = 0;
  for (int n = 0; n < NPTS; ++n) {
    float dx = pb[n] - qx, dy = pb[NPTS + n] - qy, dz = pb[2 * NPTS + n] - qz;
    float d2 = dx * dx + dy * dy + dz * dz;
    if (d2 < RAD2) {
      if (cnt == 0) first = n;
      out[cnt] = n;
      if (++cnt == KNBR) break;
    }
  }
  for (int k = cnt; k < KNBR; ++k) out[k] = first;
}

// ============================================================
// 3) Weight prep: fold BN, cast to f16, swizzle into WMMA B-fragment blobs.
//    Layer1 columns reordered: X = [feat(64) | gp(3) | pad..96]
// ============================================================
__global__ __launch_bounds__(256) void prep_kernel(
    const float* w1, const float* b1, const float* g1, const float* be1, const float* m1, const float* v1,
    const float* w2, const float* b2, const float* g2, const float* be2, const float* m2, const float* v2,
    const float* w3, const float* b3, const float* g3, const float* be3, const float* m3, const float* v3,
    _Float16* wb1, _Float16* wb2, _Float16* wb3, float* ab) {
  const int gid = blockIdx.x * blockDim.x + threadIdx.x;
  if (gid < 6144) {                       // layer1: 3 kc * 4 nt * 512
    int e = gid & 15, ln = (gid >> 4) & 31, rest = gid >> 9;
    int nt = rest & 3, kc = rest >> 2;
    int cip = kc * 32 + (ln >> 4) * 8 + (e & 7) + (e >> 3) * 16;   // permuted ci
    int co = nt * 16 + (ln & 15);
    float val = 0.0f;
    if (cip < 64)      val = w1[co * 67 + 3 + cip];       // features
    else if (cip < 67) val = w1[co * 67 + (cip - 64)];    // gp xyz
    wb1[gid] = (_Float16)val;
  } else if (gid < 10240) {               // layer2: 2 kc * 4 nt * 512
    int t = gid - 6144;
    int e = t & 15, ln = (t >> 4) & 31, rest = t >> 9;
    int nt = rest & 3, kc = rest >> 2;
    int ci = kc * 32 + (ln >> 4) * 8 + (e & 7) + (e >> 3) * 16;
    int co = nt * 16 + (ln & 15);
    wb2[t] = (_Float16)w2[co * 64 + ci];
  } else if (gid < 18432) {               // layer3: 2 kc * 8 nt * 512
    int t = gid - 10240;
    int e = t & 15, ln = (t >> 4) & 31, rest = t >> 9;
    int nt = rest & 7, kc = rest >> 3;
    int ci = kc * 32 + (ln >> 4) * 8 + (e & 7) + (e >> 3) * 16;
    int co = nt * 16 + (ln & 15);
    wb3[t] = (_Float16)w3[co * 64 + ci];
  }
  if (gid < 64) {
    float s1 = g1[gid] * rsqrtf(v1[gid] + BNEPS);
    ab[gid] = s1;            ab[64 + gid]  = (b1[gid] - m1[gid]) * s1 + be1[gid];
    float s2 = g2[gid] * rsqrtf(v2[gid] + BNEPS);
    ab[128 + gid] = s2;      ab[192 + gid] = (b2[gid] - m2[gid]) * s2 + be2[gid];
  }
  if (gid < 128) {
    float s3 = g3[gid] * rsqrtf(v3[gid] + BNEPS);
    ab[256 + gid] = s3;      ab[384 + gid] = (b3[gid] - m3[gid]) * s3 + be3[gid];
  }
}

// ============================================================
// 4) Fused gather -> MLP(67->64->64->128 via WMMA f16) -> max over K
//    One query per wave, 4 waves per block, activations in per-wave LDS.
// ============================================================
__global__ __launch_bounds__(128) void setconv_kernel(
    const float* __restrict__ points, const float* __restrict__ features,
    const int* __restrict__ fpsIdx, const int* __restrict__ nn,
    const _Float16* __restrict__ wb1, const _Float16* __restrict__ wb2,
    const _Float16* __restrict__ wb3, const float* __restrict__ ab,
    float* __restrict__ outFeat) {
  __shared__ __align__(32) _Float16 lds[4 * 5120]; // per wave: X 32x96 + Y 32x64
  const int lane = threadIdx.x & 31;
  const int wv = threadIdx.x >> 5;
  const int qid = blockIdx.x * 4 + wv;             // 0 .. B*S-1
  const int b = qid >> 11;
  const int s = qid & (SOUT - 1);
  _Float16* X = lds + wv * 5120;                   // stride 96 (f16), later reused as Y2
  _Float16* Y = X + 3072;                          // stride 64 (f16)
  const int nL = lane & 15;
  const int khalf = lane >> 4;

  // ---- build X: lane = neighbor row. cols [0..63]=feat, [64..66]=gp, [67..95]=0
  {
    const int idx = nn[(size_t)qid * KNBR + lane];
    const int cidx = fpsIdx[qid];
    const float* pb = points + (size_t)b * 3 * NPTS;
    const float gx = pb[idx] - pb[cidx];
    const float gy = pb[NPTS + idx] - pb[NPTS + cidx];
    const float gz = pb[2 * NPTS + idx] - pb[2 * NPTS + cidx];
    const float* fb = features + (size_t)b * CIN * NPTS;
    half8* xp = (half8*)(X + lane * 96);
#pragma unroll
    for (int j = 0; j < 8; ++j) {
      half8 h = {};
#pragma unroll
      for (int e = 0; e < 8; ++e) h[e] = (_Float16)fb[(size_t)(j * 8 + e) * NPTS + idx];
      xp[j] = h;
    }
    half8 hg = {};
    hg[0] = (_Float16)gx; hg[1] = (_Float16)gy; hg[2] = (_Float16)gz;
    xp[8] = hg;
    half8 hz = {};
    xp[9] = hz; xp[10] = hz; xp[11] = hz;
  }
  __syncthreads();

  // ---- layer1: X(32x96) * W1^T -> Y(32x64), BN+ReLU
  for (int mt = 0; mt < 2; ++mt) {
    const int m = mt * 16 + nL;
    const _Float16* xb = X + m * 96 + khalf * 8;
    AV a0, a1, a2;
    a0.h[0] = *(const half8*)(xb);       a0.h[1] = *(const half8*)(xb + 16);
    a1.h[0] = *(const half8*)(xb + 32);  a1.h[1] = *(const half8*)(xb + 48);
    a2.h[0] = *(const half8*)(xb + 64);  a2.h[1] = *(const half8*)(xb + 80);
    for (int nt = 0; nt < 4; ++nt) {
      v8f c = {};
      c = wmma_f16(a0.v, loadB(wb1, 0, 4, nt, lane), c);
      c = wmma_f16(a1.v, loadB(wb1, 1, 4, nt, lane), c);
      c = wmma_f16(a2.v, loadB(wb1, 2, 4, nt, lane), c);
      const float alpha = ab[nt * 16 + nL];
      const float beta  = ab[64 + nt * 16 + nL];
#pragma unroll
      for (int r = 0; r < 8; ++r) {
        float v = fmaf(c[r], alpha, beta);
        v = v > 0.0f ? v : 0.0f;
        Y[(mt * 16 + khalf * 8 + r) * 64 + nt * 16 + nL] = (_Float16)v;
      }
    }
  }
  __syncthreads();

  // ---- layer2: Y(32x64) * W2^T -> Y2 (reuse X region, stride 64)
  _Float16* Y2 = X;
  for (int mt = 0; mt < 2; ++mt) {
    const int m = mt * 16 + nL;
    const _Float16* yb = Y + m * 64 + khalf * 8;
    AV a0, a1;
    a0.h[0] = *(const half8*)(yb);       a0.h[1] = *(const half8*)(yb + 16);
    a1.h[0] = *(const half8*)(yb + 32);  a1.h[1] = *(const half8*)(yb + 48);
    for (int nt = 0; nt < 4; ++nt) {
      v8f c = {};
      c = wmma_f16(a0.v, loadB(wb2, 0, 4, nt, lane), c);
      c = wmma_f16(a1.v, loadB(wb2, 1, 4, nt, lane), c);
      const float alpha = ab[128 + nt * 16 + nL];
      const float beta  = ab[192 + nt * 16 + nL];
#pragma unroll
      for (int r = 0; r < 8; ++r) {
        float v = fmaf(c[r], alpha, beta);
        v = v > 0.0f ? v : 0.0f;
        Y2[(mt * 16 + khalf * 8 + r) * 64 + nt * 16 + nL] = (_Float16)v;
      }
    }
  }
  __syncthreads();

  // ---- layer3: Y2(32x64) * W3^T -> 32x128, BN+ReLU, max over 32 rows
  AV a[2][2];
#pragma unroll
  for (int mt = 0; mt < 2; ++mt) {
    const int m = mt * 16 + nL;
    const _Float16* yb = Y2 + m * 64 + khalf * 8;
    a[mt][0].h[0] = *(const half8*)(yb);       a[mt][0].h[1] = *(const half8*)(yb + 16);
    a[mt][1].h[0] = *(const half8*)(yb + 32);  a[mt][1].h[1] = *(const half8*)(yb + 48);
  }
  for (int nt = 0; nt < 8; ++nt) {
    const v16h b0 = loadB(wb3, 0, 8, nt, lane);
    const v16h b1 = loadB(wb3, 1, 8, nt, lane);
    const float alpha = ab[256 + nt * 16 + nL];
    const float beta  = ab[384 + nt * 16 + nL];
    float cm = 0.0f; // ReLU outputs are >= 0
#pragma unroll
    for (int mt = 0; mt < 2; ++mt) {
      v8f c = {};
      c = wmma_f16(a[mt][0].v, b0, c);
      c = wmma_f16(a[mt][1].v, b1, c);
      float lm = 0.0f;
#pragma unroll
      for (int r = 0; r < 8; ++r) {
        float v = fmaf(c[r], alpha, beta);
        v = v > 0.0f ? v : 0.0f;
        lm = fmaxf(lm, v);
      }
      cm = fmaxf(cm, lm);
    }
    cm = fmaxf(cm, __shfl_xor(cm, 16)); // combine wave halves (rows 0-7 vs 8-15)
    if (lane < 16)
      outFeat[((size_t)(b * 128 + nt * 16 + lane)) * SOUT + s] = cm;
  }
}

// ============================================================
// launch
// ============================================================
extern "C" void kernel_launch(void* const* d_in, const int* in_sizes, int n_in,
                              void* d_out, int out_size, void* d_ws, size_t ws_size,
                              hipStream_t stream) {
  const float* points   = (const float*)d_in[0];
  const float* features = (const float*)d_in[1];
  const float* w1 = (const float*)d_in[2],  *b1 = (const float*)d_in[3],
             * g1 = (const float*)d_in[4],  *be1 = (const float*)d_in[5],
             * m1 = (const float*)d_in[6],  *v1 = (const float*)d_in[7];
  const float* w2 = (const float*)d_in[8],  *b2 = (const float*)d_in[9],
             * g2 = (const float*)d_in[10], *be2 = (const float*)d_in[11],
             * m2 = (const float*)d_in[12], *v2 = (const float*)d_in[13];
  const float* w3 = (const float*)d_in[14], *b3 = (const float*)d_in[15],
             * g3 = (const float*)d_in[16], *be3 = (const float*)d_in[17],
             * m3 = (const float*)d_in[18], *v3 = (const float*)d_in[19];

  float* outPts  = (float*)d_out;                       // (B,3,S)
  float* outFeat = outPts + BATCH * 3 * SOUT;           // (B,128,S)

  char* ws = (char*)d_ws;
  int*      fpsIdx = (int*)(ws);                        // 32768 B
  int*      nnIdx  = (int*)(ws + 32768);                // 1048576 B
  _Float16* wb1    = (_Float16*)(ws + 1081344);         // 12288 B
  _Float16* wb2    = (_Float16*)(ws + 1093632);         // 8192 B
  _Float16* wb3    = (_Float16*)(ws + 1101824);         // 16384 B
  float*    ab     = (float*)(ws + 1118208);            // 2048 B

  fps_kernel<<<BATCH, 1024, 0, stream>>>(points, fpsIdx, outPts);
  prep_kernel<<<72, 256, 0, stream>>>(w1, b1, g1, be1, m1, v1,
                                      w2, b2, g2, be2, m2, v2,
                                      w3, b3, g3, be3, m3, v3,
                                      wb1, wb2, wb3, ab);
  ball_kernel<<<(BATCH * SOUT) / 256, 256, 0, stream>>>(points, fpsIdx, nnIdx);
  setconv_kernel<<<(BATCH * SOUT) / 4, 128, 0, stream>>>(points, features, fpsIdx, nnIdx,
                                                         wb1, wb2, wb3, ab, outFeat);
}
